// TimePotentialSU3_53051436040470
// MI455X (gfx1250) — matrix operand order repaired and checked by reading, not compiled, not run on hardware
//
#include <hip/hip_runtime.h>
#include <hip/hip_bf16.h>
#include <math.h>

typedef __attribute__((ext_vector_type(16))) _Float16 v16h;
typedef __attribute__((ext_vector_type(2)))  _Float16 v2h;
typedef __attribute__((ext_vector_type(8)))  float    v8f;

#define H        64
#define NB       262144
#define BQ       32          // batches per wave
#define WAVES    2
#define BLOCK    (WAVES * 32)
#define GRID     (NB / (WAVES * BQ))   // 4096

// ---------------- fast branch-free device math ----------------
__device__ __forceinline__ float ftanh(float x){
  // tanh(x) = (e - 1) / (e + 1),  e = exp(2x);  clamp keeps e finite (sat at +-1)
  float cx = fminf(fmaxf(x, -10.0f), 10.0f);
  float e  = __builtin_amdgcn_exp2f(cx * 2.8853900817779268f);  // 2*log2(e)
  return (e - 1.0f) * __builtin_amdgcn_rcpf(e + 1.0f);
}
__device__ __forceinline__ float fcbrt(float r){   // r >= 0
  return __builtin_amdgcn_exp2f(__builtin_amdgcn_logf(r) * (1.0f/3.0f));
}

// ---------------- complex helpers (split f32) ----------------
struct C2 { float x, y; };
__device__ __forceinline__ C2 mkc(float r, float i){ C2 z; z.x=r; z.y=i; return z; }
__device__ __forceinline__ C2 cadd(C2 a, C2 b){ return mkc(a.x+b.x, a.y+b.y); }
__device__ __forceinline__ C2 csub(C2 a, C2 b){ return mkc(a.x-b.x, a.y-b.y); }
__device__ __forceinline__ C2 cmul(C2 a, C2 b){ return mkc(a.x*b.x - a.y*b.y, a.x*b.y + a.y*b.x); }
__device__ __forceinline__ C2 cscale(C2 a, float s){ return mkc(a.x*s, a.y*s); }
__device__ __forceinline__ C2 cdiv(C2 a, C2 b){
  float id = __builtin_amdgcn_rcpf(b.x*b.x + b.y*b.y + 1e-30f);
  return mkc((a.x*b.x + a.y*b.y)*id, (a.y*b.x - a.x*b.y)*id);
}
__device__ __forceinline__ C2 csqrtc(C2 z){
  float r  = sqrtf(z.x*z.x + z.y*z.y);
  float re = sqrtf(fmaxf(0.5f*(r + z.x), 0.0f));
  float im = sqrtf(fmaxf(0.5f*(r - z.x), 0.0f));
  return mkc(re, z.y < 0.0f ? -im : im);
}
__device__ __forceinline__ C2 ccbrt(C2 z){   // principal complex cube root
  float r  = sqrtf(z.x*z.x + z.y*z.y);
  float m  = fcbrt(r);
  float th = atan2f(z.y, z.x) * (1.0f/3.0f);   // |th| <= pi/3: native sin/cos safe
  return mkc(m * __cosf(th), m * __sinf(th));
}

// ---------------- WMMA fragment helpers ----------------
union F4x2 { v16h v; struct { float4 a, b; } q; };

// A fragment: 16x32 f16 tile from row-major plane P[row][64], rows row0..row0+15,
// K window = 32*kt..32*kt+31.  Lanes 0-15: K base 0; lanes 16-31: K base +16.
__device__ __forceinline__ v16h loadA(const _Float16* P, int row0, int kt, int lane){
  int m  = lane & 15;
  int kb = (lane >> 4) << 4;
  const _Float16* p = P + (row0 + m) * H + (kt << 5) + kb;
  F4x2 u;
  u.q.a = *(const float4*)(p);
  u.q.b = *(const float4*)(p + 8);
  return u.v;
}

// B fragment: 32x16 f16 tile from TRANSPOSED plane Wt[n][64] (n = col, k contiguous).
__device__ __forceinline__ v16h loadB(const _Float16* Wt, int nt, int kt, int lane){
  int n  = (nt << 4) + (lane & 15);
  int kb = (kt << 5) + ((lane >> 4) << 4);
  const _Float16* p = Wt + n * H + kb;
  F4x2 u;
  u.q.a = *(const float4*)(p);
  u.q.b = *(const float4*)(p + 8);
  return u.v;
}

__device__ __forceinline__ v8f wmma_f16(v16h a, v16h b, v8f c){
  return __builtin_amdgcn_wmma_f32_16x16x32_f16(false, a, false, b, (short)0, c, false, false);
}
__device__ __forceinline__ v8f splat8(float x){
  v8f v;
#pragma unroll
  for (int i = 0; i < 8; ++i) v[i] = x;
  return v;
}
__device__ __forceinline__ v8f tanh8(v8f a){
  v8f v;
#pragma unroll
  for (int i = 0; i < 8; ++i) v[i] = ftanh(a[i]);
  return v;
}

// ---------------- kernel ----------------
__global__ __launch_bounds__(BLOCK)
void su3_deepset_kernel(const float* __restrict__ x_re, const float* __restrict__ x_im,
                        const float* __restrict__ t_in,
                        const float* __restrict__ W1r, const float* __restrict__ W1i,
                        const float* __restrict__ b1r, const float* __restrict__ b1i,
                        const float* __restrict__ W2r, const float* __restrict__ W2i,
                        const float* __restrict__ b2r, const float* __restrict__ b2i,
                        const float* __restrict__ W3r, const float* __restrict__ W3i,
                        const float* __restrict__ b3r, const float* __restrict__ b3i,
                        const float* __restrict__ W4r, const float* __restrict__ W4i,
                        const float* __restrict__ b4r, const float* __restrict__ b4i,
                        float* __restrict__ out)
{
  // LDS: weights (f16, transposed) + per-wave activation staging planes. ~51 KB.
  __shared__ _Float16 sW2r[H*H], sW2i[H*H], sW3r[H*H], sW3i[H*H];   // [n][k]
  __shared__ float sW1r[2*H], sW1i[2*H];
  __shared__ float sb1r[H], sb1i[H], sb2r[H], sb2i[H], sb3r[H], sb3i[H];
  __shared__ float sW4r[H], sW4i[H];
  __shared__ _Float16 stage[WAVES][2][BQ*H];   // [wave][re/im][row*64]

  const int tid  = threadIdx.x;
  const int wv   = tid >> 5;
  const int lane = tid & 31;

  // ---- cooperative weight load (transpose W2/W3 so B frags are contiguous) ----
  for (int i = tid; i < H*H; i += BLOCK){
    int n = i & (H-1), k = i >> 6;
    sW2r[n*H + k] = (_Float16)W2r[i];
    sW2i[n*H + k] = (_Float16)W2i[i];
    sW3r[n*H + k] = (_Float16)W3r[i];
    sW3i[n*H + k] = (_Float16)W3i[i];
  }
  for (int i = tid; i < 2*H; i += BLOCK){ sW1r[i] = W1r[i]; sW1i[i] = W1i[i]; }
  for (int i = tid; i < H; i += BLOCK){
    sb1r[i]=b1r[i]; sb1i[i]=b1i[i]; sb2r[i]=b2r[i]; sb2i[i]=b2i[i];
    sb3r[i]=b3r[i]; sb3i[i]=b3i[i]; sW4r[i]=W4r[i]; sW4i[i]=W4i[i];
  }
  __syncthreads();

  const int bb = (blockIdx.x * WAVES + wv) * BQ + lane;   // this lane's batch

  // ---- Cardano eigensolve of 3x3 complex matrix (per lane), vectorized loads ----
  const float* xr = x_re + (size_t)bb * 9;
  const float* xi = x_im + (size_t)bb * 9;
  float4 r0 = *(const float4*)(xr);     // unaligned-capable global_load_b128
  float4 r1 = *(const float4*)(xr + 4);
  float  r8 = xr[8];
  float4 i0 = *(const float4*)(xi);
  float4 i1 = *(const float4*)(xi + 4);
  float  i8 = xi[8];
  C2 Ma = mkc(r0.x,i0.x), Mb = mkc(r0.y,i0.y), Mc = mkc(r0.z,i0.z);
  C2 Md = mkc(r0.w,i0.w), Me = mkc(r1.x,i1.x), Mf = mkc(r1.y,i1.y);
  C2 Mg = mkc(r1.z,i1.z), Mh = mkc(r1.w,i1.w), Mi = mkc(r8,  i8);

  C2 m0  = csub(cmul(Me,Mi), cmul(Mf,Mh));
  C2 m1  = csub(cmul(Ma,Mi), cmul(Mc,Mg));
  C2 m2  = csub(cmul(Ma,Me), cmul(Mb,Md));
  C2 det = cadd(csub(cmul(Ma,m0), cmul(Mb, csub(cmul(Md,Mi), cmul(Mf,Mg)))),
                cmul(Mc, csub(cmul(Md,Mh), cmul(Me,Mg))));
  C2 c2  = cscale(cadd(cadd(Ma,Me),Mi), -1.0f);
  C2 c1  = cadd(cadd(m0,m1),m2);
  C2 c0  = cscale(det, -1.0f);
  C2 c2s = cmul(c2,c2);
  C2 p   = csub(c1, cscale(c2s, 1.0f/3.0f));
  C2 q   = cadd(csub(cscale(cmul(c2s,c2), 2.0f/27.0f), cscale(cmul(c2,c1), 1.0f/3.0f)), c0);
  C2 disc= csqrtc(cadd(cscale(cmul(q,q), 0.25f), cscale(cmul(cmul(p,p),p), 1.0f/27.0f)));
  C2 u   = ccbrt(cadd(cscale(q,-0.5f), disc));
  C2 wrt = mkc(-0.5f, 0.86602540378443864676f);   // exp(2*pi*i/3)
  C2 c2d3= cscale(c2, 1.0f/3.0f);
  C2 pd3 = cscale(p,  1.0f/3.0f);
  C2 eig[3];
  C2 uk = u;
#pragma unroll
  for (int k = 0; k < 3; ++k){
    eig[k] = csub(csub(uk, cdiv(pd3, uk)), c2d3);
    uk = cmul(uk, wrt);
  }
  const float tv = t_in[bb];

  // ---- pooled accumulators (fragment layout, f32) ----
  v8f sfr[2][4], sfi[2][4];
#pragma unroll
  for (int mt = 0; mt < 2; ++mt)
#pragma unroll
    for (int nt = 0; nt < 4; ++nt){ sfr[mt][nt] = splat8(0.0f); sfi[mt][nt] = splat8(0.0f); }

  _Float16* Pr = &stage[wv][0][0];
  _Float16* Pi = &stage[wv][1][0];

  // ---- layer1 (VALU) + layer2 (WMMA) per set element, pooled with tanh ----
  for (int e = 0; e < 3; ++e){
    C2 lam = eig[e];
    v2h* rowR = (v2h*)(Pr + lane * H);   // pack pairs -> ds_store_b32
    v2h* rowI = (v2h*)(Pi + lane * H);
    for (int j = 0; j < H; j += 2){
      float a0r = sW1r[j],   a0i = sW1i[j],   a1r = sW1r[H+j],   a1i = sW1i[H+j];
      float c0r = sW1r[j+1], c0i = sW1i[j+1], c1r = sW1r[H+j+1], c1i = sW1i[H+j+1];
      float h0r = lam.x*a0r - lam.y*a0i + tv*a1r + sb1r[j];
      float h0i = lam.x*a0i + lam.y*a0r + tv*a1i + sb1i[j];
      float h1r = lam.x*c0r - lam.y*c0i + tv*c1r + sb1r[j+1];
      float h1i = lam.x*c0i + lam.y*c0r + tv*c1i + sb1i[j+1];
      v2h pr; pr[0] = (_Float16)ftanh(h0r); pr[1] = (_Float16)ftanh(h1r);
      v2h pi; pi[0] = (_Float16)ftanh(h0i); pi[1] = (_Float16)ftanh(h1i);
      rowR[j >> 1] = pr;
      rowI[j >> 1] = pi;
    }
    __syncthreads();

    for (int mt = 0; mt < 2; ++mt){
      v16h Ar0 = loadA(Pr, 16*mt, 0, lane), Ar1 = loadA(Pr, 16*mt, 1, lane);
      v16h Ai0 = loadA(Pi, 16*mt, 0, lane), Ai1 = loadA(Pi, 16*mt, 1, lane);
#pragma unroll
      for (int nt = 0; nt < 4; ++nt){
        v16h Br0 = loadB(sW2r, nt, 0, lane), Br1 = loadB(sW2r, nt, 1, lane);
        v16h Bi0 = loadB(sW2i, nt, 0, lane), Bi1 = loadB(sW2i, nt, 1, lane);
        int nn = (nt << 4) + (lane & 15);
        v8f dr = splat8(sb2r[nn]);
        v8f di = splat8(sb2i[nn]);
        dr = wmma_f16(Ar1, Br1, wmma_f16(Ar0, Br0, dr));
        v8f dm = wmma_f16(Ai1, Bi1, wmma_f16(Ai0, Bi0, splat8(0.0f)));
        dr = dr - dm;                                          // Dr = ArBr - AiBi + br
        di = wmma_f16(Ai1, Br1, wmma_f16(Ai0, Br0,
             wmma_f16(Ar1, Bi1, wmma_f16(Ar0, Bi0, di))));     // Di = ArBi + AiBr + bi
        sfr[mt][nt] += tanh8(dr);
        sfi[mt][nt] += tanh8(di);
      }
    }
    __syncthreads();    // before next element overwrites the staging plane
  }

  // ---- write pooled s back to staging plane as f16 A-layout [row][k] ----
#pragma unroll
  for (int mt = 0; mt < 2; ++mt)
#pragma unroll
    for (int nt = 0; nt < 4; ++nt){
      int col = (nt << 4) + (lane & 15);
      int rb  = (mt << 4) + ((lane >> 4) << 3);
#pragma unroll
      for (int r = 0; r < 8; ++r){
        Pr[(rb + r) * H + col] = (_Float16)sfr[mt][nt][r];
        Pi[(rb + r) * H + col] = (_Float16)sfi[mt][nt][r];
      }
    }
  __syncthreads();

  // ---- layer3 (WMMA): r = tanh(s @ W3 + b3) ----
  v8f rfr[2][4], rfi[2][4];
  for (int mt = 0; mt < 2; ++mt){
    v16h Ar0 = loadA(Pr, 16*mt, 0, lane), Ar1 = loadA(Pr, 16*mt, 1, lane);
    v16h Ai0 = loadA(Pi, 16*mt, 0, lane), Ai1 = loadA(Pi, 16*mt, 1, lane);
#pragma unroll
    for (int nt = 0; nt < 4; ++nt){
      v16h Br0 = loadB(sW3r, nt, 0, lane), Br1 = loadB(sW3r, nt, 1, lane);
      v16h Bi0 = loadB(sW3i, nt, 0, lane), Bi1 = loadB(sW3i, nt, 1, lane);
      int nn = (nt << 4) + (lane & 15);
      v8f dr = splat8(sb3r[nn]);
      v8f di = splat8(sb3i[nn]);
      dr = wmma_f16(Ar1, Br1, wmma_f16(Ar0, Br0, dr));
      v8f dm = wmma_f16(Ai1, Bi1, wmma_f16(Ai0, Bi0, splat8(0.0f)));
      dr = dr - dm;
      di = wmma_f16(Ai1, Br1, wmma_f16(Ai0, Br0,
           wmma_f16(Ar1, Bi1, wmma_f16(Ar0, Bi0, di))));
      rfr[mt][nt] = tanh8(dr);
      rfi[mt][nt] = tanh8(di);
    }
  }
  __syncthreads();

  // ---- stage r, then layer4 (complex matvec, VALU) ----
#pragma unroll
  for (int mt = 0; mt < 2; ++mt)
#pragma unroll
    for (int nt = 0; nt < 4; ++nt){
      int col = (nt << 4) + (lane & 15);
      int rb  = (mt << 4) + ((lane >> 4) << 3);
#pragma unroll
      for (int r = 0; r < 8; ++r){
        Pr[(rb + r) * H + col] = (_Float16)rfr[mt][nt][r];
        Pi[(rb + r) * H + col] = (_Float16)rfi[mt][nt][r];
      }
    }
  __syncthreads();

  float or_ = b4r[0], oi_ = b4i[0];
  for (int k = 0; k < H; ++k){
    float rr = (float)Pr[lane * H + k];
    float ri = (float)Pi[lane * H + k];
    float wr4 = sW4r[k], wi4 = sW4i[k];
    or_ += rr*wr4 - ri*wi4;
    oi_ += rr*wi4 + ri*wr4;
  }
  float2 o; o.x = or_; o.y = oi_;
  *(float2*)(out + 2*(size_t)bb) = o;   // complex64 interleaved, one b64 store
}

extern "C" void kernel_launch(void* const* d_in, const int* in_sizes, int n_in,
                              void* d_out, int out_size, void* d_ws, size_t ws_size,
                              hipStream_t stream) {
  const float* x_re = (const float*)d_in[0];
  const float* x_im = (const float*)d_in[1];
  const float* t    = (const float*)d_in[2];
  const float* W1r  = (const float*)d_in[3];
  const float* W1i  = (const float*)d_in[4];
  const float* b1r  = (const float*)d_in[5];
  const float* b1i  = (const float*)d_in[6];
  const float* W2r  = (const float*)d_in[7];
  const float* W2i  = (const float*)d_in[8];
  const float* b2r  = (const float*)d_in[9];
  const float* b2i  = (const float*)d_in[10];
  const float* W3r  = (const float*)d_in[11];
  const float* W3i  = (const float*)d_in[12];
  const float* b3r  = (const float*)d_in[13];
  const float* b3i  = (const float*)d_in[14];
  const float* W4r  = (const float*)d_in[15];
  const float* W4i  = (const float*)d_in[16];
  const float* b4r  = (const float*)d_in[17];
  const float* b4i  = (const float*)d_in[18];
  float* out = (float*)d_out;

  su3_deepset_kernel<<<GRID, BLOCK, 0, stream>>>(
      x_re, x_im, t, W1r, W1i, b1r, b1i, W2r, W2i, b2r, b2i,
      W3r, W3i, b3r, b3i, W4r, W4i, b4r, b4i, out);
}
